// LearntNeighbourhoodSampling_59768764891572
// MI455X (gfx1250) — matrix operand order repaired and checked by reading, not compiled, not run on hardware
//
#include <hip/hip_runtime.h>

typedef _Float16 h8   __attribute__((ext_vector_type(8)));
typedef _Float16 v16h __attribute__((ext_vector_type(16)));
typedef float    v8f  __attribute__((ext_vector_type(8)));

#define NVERT 16000
#define C_IN  256
#define C3    259          // C + 3 coord channels
#define KNB   27
#define DV 16
#define HV 16
#define WV 11
#define NVOX (DV*HV*WV)    // 2816
#define TILE_N 32
#define CKPAD 288          // 259 padded to 9 chunks of 32
#define NKCH  9

// ---- workspace layout (bytes) ----
#define OFF_VOLT   0                                  // f32 [2816][256]
#define OFF_G      (OFF_VOLT + NVOX*C_IN*4)           // f32 [256][259]  G = W_d2 @ W_d1
#define OFF_WC     (OFF_G    + C_IN*C3*4)             // f32 [256][259]  Wc = W_c2 @ W_c1
#define OFF_WSS    (OFF_WC   + C_IN*C3*4)             // f32 [256][256]  sum_k W_sum_k
#define OFF_VB     (OFF_WSS  + C_IN*C_IN*4)           // f32 [256]       W_d2@b_d1 + b_d2
#define OFF_CTOT   (OFF_VB   + C_IN*4)                // f32 [256]
#define OFF_PACKB  (OFF_CTOT + C_IN*4)                // f16 [27][9][16][32][16]
#define PACKB_HALVES (KNB*NKCH*16*32*16)              // 1,990,656 halves (~4MB)

// =====================================================================
// Precompute kernels (all tiny vs. the main GEMM)
// =====================================================================

// vol [C][D][H][W] -> volT [vox][C] (channel-minor for coalesced gathers)
__global__ void k_transpose(const float* __restrict__ vol, float* __restrict__ volT) {
    int vox = blockIdx.x;          // 0..2815
    int c   = threadIdx.x;         // 0..255
    volT[vox * C_IN + c] = vol[c * NVOX + vox];
}

// G = W_d2 @ W_d1, Wc = W_c2 @ W_c1   (both [256][259])
__global__ void k_gwc(const float* __restrict__ W_d1, const float* __restrict__ W_d2,
                      const float* __restrict__ W_c1, const float* __restrict__ W_c2,
                      float* __restrict__ G, float* __restrict__ Wc) {
    int idx = blockIdx.x * 256 + threadIdx.x;
    int total = C_IN * C3;
    if (idx >= 2 * total) return;
    int which = idx >= total;
    int i2 = which ? idx - total : idx;
    int o = i2 / C3, ck = i2 % C3;
    const float* A = which ? W_c2 : W_d2;
    const float* B = which ? W_c1 : W_d1;
    float acc = 0.f;
    for (int i = 0; i < C_IN; ++i)
        acc += A[o * C_IN + i] * B[i * C3 + ck];
    (which ? Wc : G)[i2] = acc;
}

// Wss[p][o] = sum_k W_sum[p][o][k];  vb[o] = sum_i W_d2[o][i]*b_d1[i] + b_d2[o]
__global__ void k_sumvb(const float* __restrict__ W_sum, const float* __restrict__ W_d2,
                        const float* __restrict__ b_d1, const float* __restrict__ b_d2,
                        float* __restrict__ Wss, float* __restrict__ vb) {
    int idx = blockIdx.x * 256 + threadIdx.x;
    if (idx < C_IN * C_IN) {
        float s = 0.f;
        const float* w = W_sum + (size_t)idx * KNB;
        for (int k = 0; k < KNB; ++k) s += w[k];
        Wss[idx] = s;
    } else if (idx < C_IN * C_IN + C_IN) {
        int o = idx - C_IN * C_IN;
        float s = b_d2[o];
        for (int i = 0; i < C_IN; ++i) s += W_d2[o * C_IN + i] * b_d1[i];
        vb[o] = s;
    }
}

// cTot[p] = b_sum[p] + sum_o Wss[p][o]*vb[o] + sum_o W_c2[p][o]*b_c1[o] + b_c2[p]
__global__ void k_ctot(const float* __restrict__ Wss, const float* __restrict__ vb,
                       const float* __restrict__ b_sum, const float* __restrict__ W_c2,
                       const float* __restrict__ b_c1, const float* __restrict__ b_c2,
                       float* __restrict__ cTot) {
    int p = threadIdx.x;
    float s = b_sum[p] + b_c2[p];
    for (int o = 0; o < C_IN; ++o) {
        s += Wss[p * C_IN + o] * vb[o];
        s += W_c2[p * C_IN + o] * b_c1[o];
    }
    cTot[p] = s;
}

// Pack folded weights directly in WMMA B-fragment order:
// idx = (((k*9+kk)*16 + ntile)*32 + lane)*16 + j
//   col p = ntile*16 + (lane&15);  K-in-chunk = (lane>>4)*8 + (j&7) + (j>>3)*16;  ck = kk*32 + Kin
// value = Ã_k[p][ck] where Ã_k = W_sum_k@G (+ corrections k==0: -Wss@G, k==13: +Wc)
__global__ void k_packb(const float* __restrict__ W_sum, const float* __restrict__ G,
                        const float* __restrict__ Wss, const float* __restrict__ Wc,
                        _Float16* __restrict__ packB) {
    int idx = blockIdx.x * 256 + threadIdx.x;
    if (idx >= PACKB_HALVES) return;
    int j = idx & 15;
    int r = idx >> 4;
    int lane = r & 31;  r >>= 5;
    int ntile = r & 15; r >>= 4;
    int kk = r % NKCH;
    int k  = r / NKCH;
    int p  = ntile * 16 + (lane & 15);
    int ck = kk * 32 + ((lane >> 4) << 3) + (j & 7) + ((j >> 3) << 4);
    float w = 0.f;
    if (ck < C3) {
        for (int o = 0; o < C_IN; ++o) {
            float ws = W_sum[((size_t)p * C_IN + o) * KNB + k];
            if (k == 0) ws -= Wss[p * C_IN + o];
            w += ws * G[o * C3 + ck];
        }
        if (k == 13) w += Wc[p * C3 + ck];
    }
    packB[idx] = (_Float16)w;
}

// =====================================================================
// Fused main kernel: sample -> folded GEMM via WMMA f32_16x16x32_f16
// =====================================================================

__device__ inline float tri_sample(const float* __restrict__ volT,
                                   float cx, float cy, float cz, int c) {
    float x = fminf(fmaxf((cx + 1.f) * 5.0f, 0.f), 10.f);   // (W-1)/2 = 5
    float y = fminf(fmaxf((cy + 1.f) * 7.5f, 0.f), 15.f);   // (H-1)/2 = 7.5
    float z = fminf(fmaxf((cz + 1.f) * 7.5f, 0.f), 15.f);   // (D-1)/2 = 7.5
    float xf = floorf(x), yf = floorf(y), zf = floorf(z);
    int x0 = (int)xf, y0 = (int)yf, z0 = (int)zf;
    int x1 = min(x0 + 1, WV - 1), y1 = min(y0 + 1, HV - 1), z1 = min(z0 + 1, DV - 1);
    float wx = x - xf, wy = y - yf, wz = z - zf;
    float ux = 1.f - wx, uy = 1.f - wy, uz = 1.f - wz;
#define AT(zz, yy, xx) volT[(size_t)(((zz) * HV + (yy)) * WV + (xx)) * C_IN + c]
    float v = 0.f;
    v += uz * uy * ux * AT(z0, y0, x0);
    v += uz * uy * wx * AT(z0, y0, x1);
    v += uz * wy * ux * AT(z0, y1, x0);
    v += uz * wy * wx * AT(z0, y1, x1);
    v += wz * uy * ux * AT(z1, y0, x0);
    v += wz * uy * wx * AT(z1, y0, x1);
    v += wz * wy * ux * AT(z1, y1, x0);
    v += wz * wy * wx * AT(z1, y1, x1);
#undef AT
    return v;
}

__global__ __launch_bounds__(256) void k_main(
    const float* __restrict__ volT, const float* __restrict__ verts,
    const float* __restrict__ W_sd, const float* __restrict__ b_sd,
    const _Float16* __restrict__ packB, const float* __restrict__ cTot,
    float* __restrict__ out) {

    __shared__ float    sVert[TILE_N][3];
    __shared__ _Float16 sFeatC[TILE_N][C_IN];
    __shared__ float    sNeigh[TILE_N][KNB][3];
    __shared__ _Float16 sX[TILE_N][CKPAD];

    const int t = threadIdx.x;
    const int n0 = blockIdx.x * TILE_N;

    if (t < TILE_N * 3) ((float*)sVert)[t] = verts[n0 * 3 + t];
    __syncthreads();

    // ---- center features (== neighbor 0 samples), f16 in LDS ----
    {
        const int c = t;
        for (int n = 0; n < TILE_N; ++n) {
            float v = tri_sample(volT, sVert[n][0], sVert[n][1], sVert[n][2], c);
            sFeatC[n][c] = (_Float16)v;
        }
    }
    __syncthreads();

    // ---- learned offsets -> neighbor coords ----
    for (int i = t; i < TILE_N * KNB; i += 256) {
        int n = i / KNB, k = i % KNB;
        float s0, s1, s2;
        if (k == 0) { s0 = s1 = s2 = 0.f; }
        else {
            s0 = b_sd[k * 3 + 0]; s1 = b_sd[k * 3 + 1]; s2 = b_sd[k * 3 + 2];
            const float* w0 = W_sd + (size_t)(k * 3 + 0) * C_IN;
            const float* w1 = W_sd + (size_t)(k * 3 + 1) * C_IN;
            const float* w2 = W_sd + (size_t)(k * 3 + 2) * C_IN;
            for (int o = 0; o < C_IN; ++o) {
                float f = (float)sFeatC[n][o];
                s0 += w0[o] * f; s1 += w1[o] * f; s2 += w2[o] * f;
            }
        }
        sNeigh[n][k][0] = sVert[n][0] + s0;
        sNeigh[n][k][1] = sVert[n][1] + s1;
        sNeigh[n][k][2] = sVert[n][2] + s2;
    }
    __syncthreads();

    // ---- wave tiling: 8 waves -> 2 M-tiles x (4 groups of 4 N-tiles) ----
    const int w = t >> 5, lane = t & 31;
    const int mtile = w & 1;
    const int ntbase = (w >> 1) * 4;
    v8f acc[4];
#pragma unroll
    for (int a = 0; a < 4; ++a) acc[a] = (v8f){0.f, 0.f, 0.f, 0.f, 0.f, 0.f, 0.f, 0.f};

    for (int k = 0; k < KNB; ++k) {
        // sample X tile [32 verts][288 ch] for this neighbor
        {
            const int c = t;
            for (int n = 0; n < TILE_N; ++n) {
                float v = tri_sample(volT, sNeigh[n][k][0], sNeigh[n][k][1],
                                     sNeigh[n][k][2], c);
                sX[n][c] = (_Float16)v;
            }
            if (t < CKPAD - C_IN) {   // pad cols 256..287: 3 raw coords then zeros
                for (int n = 0; n < TILE_N; ++n)
                    sX[n][C_IN + t] = (t < 3) ? (_Float16)sNeigh[n][k][t] : (_Float16)0.f;
            }
        }
        __syncthreads();

        const _Float16* bk = packB + (size_t)k * (NKCH * 16 * 32 * 16);
        __builtin_prefetch(bk, 0, 1);   // -> global_prefetch_b8

#pragma unroll
        for (int kk = 0; kk < NKCH; ++kk) {
            // A fragment per ISA 16-bit A 16x32 layout:
            // lanes 0-15 (M=lane):   elems 0..7 = K+0..7,  elems 8..15 = K+16..23
            // lanes 16-31 (M=lane-16): elems 0..7 = K+8..15, elems 8..15 = K+24..31
            int mrow = mtile * 16 + (lane & 15);
            int kb = kk * 32 + ((lane >> 4) << 3);
            h8 alo = *(const h8*)&sX[mrow][kb];
            h8 ahi = *(const h8*)&sX[mrow][kb + 16];
            v16h afrag;
#pragma unroll
            for (int j = 0; j < 8; ++j) { afrag[j] = alo[j]; afrag[j + 8] = ahi[j]; }
#pragma unroll
            for (int a = 0; a < 4; ++a) {
                const _Float16* bp =
                    bk + ((size_t)((kk * 16 + ntbase + a) * 32 + lane) << 4);
                v16h bfrag = *(const v16h*)bp;
                acc[a] = __builtin_amdgcn_wmma_f32_16x16x32_f16(
                    false, afrag, false, bfrag, (short)0, acc[a], false, false);
            }
        }
        __syncthreads();
    }

    // ---- epilogue: C/D layout (VGPR r: lanes0-15 M=r, lanes16-31 M=r+8) ----
#pragma unroll
    for (int a = 0; a < 4; ++a) {
        int col = (ntbase + a) * 16 + (lane & 15);
        float bias = cTot[col];
#pragma unroll
        for (int r = 0; r < 8; ++r) {
            int m = r + ((lane >> 4) << 3);
            out[(size_t)(n0 + mtile * 16 + m) * C_IN + col] = acc[a][r] + bias;
        }
    }
}

// =====================================================================

extern "C" void kernel_launch(void* const* d_in, const int* in_sizes, int n_in,
                              void* d_out, int out_size, void* d_ws, size_t ws_size,
                              hipStream_t stream) {
    const float* vol   = (const float*)d_in[0];
    const float* verts = (const float*)d_in[1];
    const float* W_sd  = (const float*)d_in[2];
    const float* b_sd  = (const float*)d_in[3];
    const float* W_d1  = (const float*)d_in[4];
    const float* b_d1  = (const float*)d_in[5];
    const float* W_d2  = (const float*)d_in[6];
    const float* b_d2  = (const float*)d_in[7];
    const float* W_c1  = (const float*)d_in[8];
    const float* b_c1  = (const float*)d_in[9];
    const float* W_c2  = (const float*)d_in[10];
    const float* b_c2  = (const float*)d_in[11];
    const float* W_sum = (const float*)d_in[12];
    const float* b_sum = (const float*)d_in[13];

    char* ws = (char*)d_ws;
    float*    volT  = (float*)(ws + OFF_VOLT);
    float*    G     = (float*)(ws + OFF_G);
    float*    Wc    = (float*)(ws + OFF_WC);
    float*    Wss   = (float*)(ws + OFF_WSS);
    float*    vb    = (float*)(ws + OFF_VB);
    float*    cTot  = (float*)(ws + OFF_CTOT);
    _Float16* packB = (_Float16*)(ws + OFF_PACKB);

    k_transpose<<<NVOX, 256, 0, stream>>>(vol, volT);
    k_gwc<<<(2 * C_IN * C3 + 255) / 256, 256, 0, stream>>>(W_d1, W_d2, W_c1, W_c2, G, Wc);
    k_sumvb<<<(C_IN * C_IN + C_IN + 255) / 256, 256, 0, stream>>>(W_sum, W_d2, b_d1, b_d2, Wss, vb);
    k_ctot<<<1, 256, 0, stream>>>(Wss, vb, b_sum, W_c2, b_c1, b_c2, cTot);
    k_packb<<<(PACKB_HALVES + 255) / 256, 256, 0, stream>>>(W_sum, G, Wss, Wc, packB);
    k_main<<<NVERT / TILE_N, 256, 0, stream>>>(volT, verts, W_sd, b_sd, packB, cTot,
                                               (float*)d_out);
}
